// SerializedAttention_51367808860480
// MI455X (gfx1250) — compile-verified, matrix operand order checked by reading, not run on hardware
//
#include <hip/hip_runtime.h>
#include <stdint.h>

// ---------------- problem constants ----------------
#define NPTS   65536
#define CCH    512
#define OC3    1536
#define NH     8
#define KPATCH 128
#define HD     64
#define NPATCH 512          // NPTS / KPATCH
#define POSB   15
#define RPEN   31           // 2*POSB+1
#define SCALE  0.125f       // HD^-0.5 = 1/8

// ---------------- vector types ----------------
typedef __attribute__((ext_vector_type(16))) __bf16   v16bf;
typedef __attribute__((ext_vector_type(8)))  float    v8f;
typedef __attribute__((ext_vector_type(2)))  float    v2f;
typedef __attribute__((ext_vector_type(4)))  uint32_t u32x4;
typedef __attribute__((ext_vector_type(8)))  uint32_t u32x8;

// ---------------- WMMA wrappers ----------------
__device__ __forceinline__ v8f wmma_bf16(v16bf a, v16bf b, v8f c) {
  return __builtin_amdgcn_wmma_f32_16x16x32_bf16(false, a, false, b,
                                                 (short)0, c, false, false);
}
__device__ __forceinline__ v8f wmma_f32k4(v2f a, v2f b, v8f c) {
  return __builtin_amdgcn_wmma_f32_16x16x4_f32(false, a, false, b,
                                               (short)0, c, false, false);
}

// bf16 16x32 A-fragment (or mirrored 32x16 B-fragment) from a row-major
// tile origin with row stride `ld` (elements). ISA layout:
//   lanes 0-15 : VGPR0-3 = K 0..7,  VGPR4-7 = K 16..23  (row = lane)
//   lanes 16-31: VGPR0-3 = K 8..15, VGPR4-7 = K 24..31  (row = lane-16)
__device__ __forceinline__ v16bf load_bf16_frag(const __bf16* tile, int ld) {
  int lane = threadIdx.x & 31;
  int r    = lane & 15;
  int lh   = lane >> 4;
  const __bf16* row = tile + (size_t)r * ld + lh * 8;
  u32x4 lo = *(const u32x4*)(row);
  u32x4 hi = *(const u32x4*)(row + 16);
  u32x8 d  = {lo[0], lo[1], lo[2], lo[3], hi[0], hi[1], hi[2], hi[3]};
  return __builtin_bit_cast(v16bf, d);
}

// ---------------- kernel 1: fp32 -> bf16 convert ----------------
__global__ void f32_to_bf16_kernel(const float* __restrict__ src,
                                   __bf16* __restrict__ dst, int n) {
  int i = blockIdx.x * blockDim.x + threadIdx.x;
  int stride = gridDim.x * blockDim.x;
  for (; i < n; i += stride) dst[i] = (__bf16)src[i];
}

// ---------------- kernel 2: QKV GEMM + gather-scatter ----------------
// Y[n,o] = sum_c feat[n,c] * W[o,c] + b[o], then row n lands at gathered
// position g = inverse[n]; column o = s*512 + h*64 + d  -> {q,k,v}buf[p][h][r][d]
__global__ __launch_bounds__(256) void qkv_gemm_scatter(
    const __bf16* __restrict__ featb,      // [N][512] bf16
    const __bf16* __restrict__ wb,         // [1536][512] bf16
    const float*  __restrict__ bias,       // [1536]
    const int*    __restrict__ inverse,    // [N]
    float* __restrict__ qbuf, float* __restrict__ kbuf, float* __restrict__ vbuf)
{
  const int rowBlk = blockIdx.x * 128;
  const int colBlk = blockIdx.y * 128;
  const int wave = threadIdx.x >> 5;
  const int lane = threadIdx.x & 31;
  const int wr = (wave & 3) * 32;   // 4 wave-rows of 32
  const int wc = (wave >> 2) * 64;  // 2 wave-cols of 64
  const int lh = lane >> 4, nidx = lane & 15;

  v8f acc[2][4] = {};
#pragma unroll 4
  for (int k0 = 0; k0 < CCH; k0 += 32) {
    v16bf afr[2], bfr[4];
#pragma unroll
    for (int r = 0; r < 2; ++r)
      afr[r] = load_bf16_frag(featb + (size_t)(rowBlk + wr + r * 16) * CCH + k0, CCH);
#pragma unroll
    for (int c = 0; c < 4; ++c)
      bfr[c] = load_bf16_frag(wb + (size_t)(colBlk + wc + c * 16) * CCH + k0, CCH);
#pragma unroll
    for (int r = 0; r < 2; ++r)
#pragma unroll
      for (int c = 0; c < 4; ++c)
        acc[r][c] = wmma_bf16(afr[r], bfr[c], acc[r][c]);
  }

#pragma unroll
  for (int r = 0; r < 2; ++r) {
    int gv[8];
#pragma unroll
    for (int v = 0; v < 8; ++v)
      gv[v] = inverse[rowBlk + wr + r * 16 + v + lh * 8];
#pragma unroll
    for (int c = 0; c < 4; ++c) {
      int o = colBlk + wc + c * 16 + nidx;
      float bo = bias[o];
      int s = o >> 9;
      int h = (o >> 6) & 7;
      int d = o & 63;
      float* dst = (s == 0) ? qbuf : (s == 1) ? kbuf : vbuf;
#pragma unroll
      for (int v = 0; v < 8; ++v) {
        int g = gv[v];
        int p = g >> 7, rr = g & 127;
        dst[(size_t)(((p * NH + h) * KPATCH) + rr) * HD + d] = acc[r][c][v] + bo;
      }
    }
  }
}

// ---------------- kernel 3: attention per (patch, head) ----------------
__global__ __launch_bounds__(256) void attn_kernel(
    const float* __restrict__ qbuf, const float* __restrict__ kbuf,
    const float* __restrict__ vbuf,
    const int* __restrict__ grid_coord,    // [N][3]
    const int* __restrict__ order,         // [N]
    const float* __restrict__ rpe_table,   // [93][8]
    __bf16* __restrict__ ctx)              // [N][512] bf16, final row order
{
  const int p = blockIdx.x;
  const int h = blockIdx.y;
  __shared__ float S[KPATCH][132];         // score matrix, padded pitch
  __shared__ int   gx[KPATCH], gy[KPATCH], gz[KPATCH];
  __shared__ float rpe[96];                // 93 used: axis*31 + idx

  const int tid = threadIdx.x;
  const int wave = tid >> 5;
  const int lane = tid & 31;
  const int lh = lane >> 4, nidx = lane & 15;

  for (int i = tid; i < KPATCH; i += 256) {
    int src = order[p * KPATCH + i];
    gx[i] = grid_coord[src * 3 + 0];
    gy[i] = grid_coord[src * 3 + 1];
    gz[i] = grid_coord[src * 3 + 2];
  }
  for (int t = tid; t < 3 * RPEN; t += 256) rpe[t] = rpe_table[t * NH + h];
  __syncthreads();

  const size_t base = (size_t)(p * NH + h) * KPATCH * HD;
  const float* qp = qbuf + base;   // [128][64]
  const float* kp = kbuf + base;
  const float* vp = vbuf + base;

  // ---- S = (q * SCALE) k^T : wave owns rows [16w, 16w+16), 8 col tiles ----
  const int iBase = wave * 16;
  v8f acc[8] = {};
  for (int d0 = 0; d0 < HD; d0 += 4) {
    v2f a = *(const v2f*)(qp + (size_t)(iBase + nidx) * HD + d0 + lh * 2);
#pragma unroll
    for (int c = 0; c < 8; ++c) {
      v2f b = *(const v2f*)(kp + (size_t)(c * 16 + nidx) * HD + d0 + lh * 2);
      acc[c] = wmma_f32k4(a, b, acc[c]);
    }
  }
  // ---- add RPE bias, scale, spill to LDS ----
#pragma unroll
  for (int c = 0; c < 8; ++c) {
#pragma unroll
    for (int v = 0; v < 8; ++v) {
      int i = iBase + v + lh * 8;
      int j = c * 16 + nidx;
      int rx = gx[i] - gx[j]; rx = rx < -POSB ? -POSB : (rx > POSB ? POSB : rx);
      int ry = gy[i] - gy[j]; ry = ry < -POSB ? -POSB : (ry > POSB ? POSB : ry);
      int rz = gz[i] - gz[j]; rz = rz < -POSB ? -POSB : (rz > POSB ? POSB : rz);
      float bias = rpe[rx + POSB] + rpe[ry + POSB + RPEN] + rpe[rz + POSB + 2 * RPEN];
      S[i][j] = acc[c][v] * SCALE + bias;
    }
  }
  __syncthreads();

  // ---- fp32 softmax: one thread per row ----
  if (tid < KPATCH) {
    float m = -3.4e38f;
    for (int j = 0; j < KPATCH; ++j) m = fmaxf(m, S[tid][j]);
    float sum = 0.f;
    for (int j = 0; j < KPATCH; ++j) {
      float e = __expf(S[tid][j] - m);
      S[tid][j] = e;
      sum += e;
    }
    float r = 1.f / sum;
    for (int j = 0; j < KPATCH; ++j) S[tid][j] *= r;
  }
  __syncthreads();

  // ---- O = P @ V : wave rows [16w,16w+16), 4 col tiles of d ----
  v8f oacc[4] = {};
  for (int j0 = 0; j0 < KPATCH; j0 += 4) {
    v2f a = *(const v2f*)&S[iBase + nidx][j0 + lh * 2];     // ds_load_b64
#pragma unroll
    for (int c = 0; c < 4; ++c) {
      const float* vv = vp + (size_t)(j0 + lh * 2) * HD + c * 16 + nidx;
      v2f b;
      b[0] = vv[0];
      b[1] = vv[HD];
      oacc[c] = wmma_f32k4(a, b, oacc[c]);
    }
  }

  // ---- scatter to final order as bf16: ctx[order[g]][h*64+d] ----
#pragma unroll
  for (int v = 0; v < 8; ++v) {
    int i = iBase + v + lh * 8;
    int n = order[p * KPATCH + i];
#pragma unroll
    for (int c = 0; c < 4; ++c) {
      int d = c * 16 + nidx;
      ctx[(size_t)n * CCH + h * HD + d] = (__bf16)oacc[c][v];
    }
  }
}

// ---------------- kernel 4: output projection GEMM ----------------
__global__ __launch_bounds__(256) void proj_gemm(
    const __bf16* __restrict__ ctxb,       // [N][512] bf16
    const __bf16* __restrict__ wb,         // [512][512] bf16
    const float*  __restrict__ bias,       // [512]
    float* __restrict__ out)               // [N][512]
{
  const int rowBlk = blockIdx.x * 128;
  const int colBlk = blockIdx.y * 128;
  const int wave = threadIdx.x >> 5;
  const int lane = threadIdx.x & 31;
  const int wr = (wave & 3) * 32;
  const int wc = (wave >> 2) * 64;
  const int lh = lane >> 4, nidx = lane & 15;

  v8f acc[2][4] = {};
#pragma unroll 4
  for (int k0 = 0; k0 < CCH; k0 += 32) {
    v16bf afr[2], bfr[4];
#pragma unroll
    for (int r = 0; r < 2; ++r)
      afr[r] = load_bf16_frag(ctxb + (size_t)(rowBlk + wr + r * 16) * CCH + k0, CCH);
#pragma unroll
    for (int c = 0; c < 4; ++c)
      bfr[c] = load_bf16_frag(wb + (size_t)(colBlk + wc + c * 16) * CCH + k0, CCH);
#pragma unroll
    for (int r = 0; r < 2; ++r)
#pragma unroll
      for (int c = 0; c < 4; ++c)
        acc[r][c] = wmma_bf16(afr[r], bfr[c], acc[r][c]);
  }

#pragma unroll
  for (int r = 0; r < 2; ++r) {
#pragma unroll
    for (int c = 0; c < 4; ++c) {
      int o = colBlk + wc + c * 16 + nidx;
      float bo = bias[o];
#pragma unroll
      for (int v = 0; v < 8; ++v) {
        int n = rowBlk + wr + r * 16 + v + lh * 8;
        out[(size_t)n * CCH + o] = acc[r][c][v] + bo;
      }
    }
  }
}

// ---------------- host launch ----------------
extern "C" void kernel_launch(void* const* d_in, const int* in_sizes, int n_in,
                              void* d_out, int out_size, void* d_ws, size_t ws_size,
                              hipStream_t stream) {
  const float* feat       = (const float*)d_in[0];
  const int*   grid_coord = (const int*)d_in[1];
  const int*   order      = (const int*)d_in[2];
  const int*   inverse    = (const int*)d_in[3];
  const float* qkv_w      = (const float*)d_in[4];
  const float* qkv_b      = (const float*)d_in[5];
  const float* proj_w     = (const float*)d_in[6];
  const float* proj_b     = (const float*)d_in[7];
  const float* rpe_table  = (const float*)d_in[8];
  float* out = (float*)d_out;

  char* ws = (char*)d_ws;
  size_t off = 0;
  __bf16* qkvw_b = (__bf16*)(ws + off); off += (size_t)OC3 * CCH * 2;
  __bf16* projw_b = (__bf16*)(ws + off); off += (size_t)CCH * CCH * 2;
  __bf16* featb  = (__bf16*)(ws + off); off += (size_t)NPTS * CCH * 2;
  float* qbuf = (float*)(ws + off); off += (size_t)NPTS * CCH * 4;
  float* kbuf = (float*)(ws + off); off += (size_t)NPTS * CCH * 4;
  float* vbuf = (float*)(ws + off); off += (size_t)NPTS * CCH * 4;
  __bf16* ctx = (__bf16*)(ws + off); off += (size_t)NPTS * CCH * 2;

  // 1) bf16 conversions
  f32_to_bf16_kernel<<<8192, 256, 0, stream>>>(feat, featb, NPTS * CCH);
  f32_to_bf16_kernel<<<1024, 256, 0, stream>>>(qkv_w, qkvw_b, OC3 * CCH);
  f32_to_bf16_kernel<<<512, 256, 0, stream>>>(proj_w, projw_b, CCH * CCH);

  // 2) QKV projection, fused gather by serialization order
  qkv_gemm_scatter<<<dim3(NPTS / 128, OC3 / 128), 256, 0, stream>>>(
      featb, qkvw_b, qkv_b, inverse, qbuf, kbuf, vbuf);

  // 3) attention per (patch, head), fused RPE + softmax + scatter-back
  attn_kernel<<<dim3(NPATCH, NH), 256, 0, stream>>>(
      qbuf, kbuf, vbuf, grid_coord, order, rpe_table, ctx);

  // 4) output projection
  proj_gemm<<<dim3(NPTS / 128, CCH / 128), 256, 0, stream>>>(
      ctx, projw_b, proj_b, out);
}
